// VisionTransformer_47150150975695
// MI455X (gfx1250) — compile-verified, matrix operand order checked by reading, not run on hardware
//
#include <hip/hip_runtime.h>
#include <hip/hip_bf16.h>
#include <math.h>

// ---------------------------------------------------------------------------
// CDNA5 (gfx1250) implementation: everything GEMM-shaped runs through a
// wave32 WMMA f16->f32 kernel (v_wmma_f32_16x16x32_f16), 64x64 block tile,
// 8 waves x (16x32) sub-tiles, double-buffered ASYNC global->LDS staging
// (GLOBAL_LOAD_ASYNC_TO_LDS_B128 + s_wait_asynccnt), fused bias/activation.
// ---------------------------------------------------------------------------

typedef _Float16 half8  __attribute__((ext_vector_type(8)));
typedef _Float16 half16 __attribute__((ext_vector_type(16)));
typedef float    float8 __attribute__((ext_vector_type(8)));
typedef int      int4v  __attribute__((ext_vector_type(4)));

#define NB   8          // batch
#define DIM  384
#define NHD  6          // heads
#define DHD  64         // head dim
#define NT0  256        // tokens stage 0
#define NT1  448        // tokens stage 1
#define KSPL 64
#define KKEP 192

// d_out layout (floats)
#define OFF_X      0
#define CNT_X      (NB*NT1*DIM)          // 1376256
#define OFF_COORDS (OFF_X + CNT_X)       // 1376256
#define CNT_COORDS (NB*NT1*3)            // 10752
#define OFF_PRED   (OFF_COORDS + CNT_COORDS)
#define CNT_PRED   (NB*NT0)              // 2048
#define OFF_META   (OFF_PRED + CNT_PRED)
#define CNT_META   (NB*NT0*2)            // 4096

// -------------------- CDNA5 async global->LDS helpers -----------------------

#if __has_builtin(__builtin_amdgcn_global_load_async_to_lds_b128)
#define HAS_ASYNC 1
#else
#define HAS_ASYNC 0
#endif

typedef __attribute__((address_space(1))) int4v* gptr_i4;
typedef __attribute__((address_space(3))) int4v* lptr_i4;

__device__ __forceinline__ void g2l_b128(const _Float16* src, _Float16* lds_dst) {
#if HAS_ASYNC
  // generic->AS1 keeps the full 64-bit address; generic LDS addr low 32 bits
  // are the LDS offset (aperture rule), so integer detours give typed AS ptrs.
  __builtin_amdgcn_global_load_async_to_lds_b128(
      (gptr_i4)(uintptr_t)src,
      (lptr_i4)(unsigned int)(uintptr_t)lds_dst,
      0, 0);
#else
  *(half8*)lds_dst = *(const half8*)src;
#endif
}

template <int N>
__device__ __forceinline__ void wait_async() {
#if HAS_ASYNC
#if __has_builtin(__builtin_amdgcn_s_wait_asynccnt)
  __builtin_amdgcn_s_wait_asynccnt((short)N);
#else
  asm volatile("s_wait_asynccnt %0" :: "n"(N) : "memory");
#endif
#endif
}

// ----------------------------- GEMM ----------------------------------------

struct GemmArgs {
  const _Float16* A; long lda; long zA0, zA1;
  const _Float16* B; long ldb; long zB0, zB1;
  const float* bias;
  float* Cf; _Float16* Ch; long ldc; long zC0, zC1;
  int M, N, K;
  int bmode;   // 0: B is [n][k] row-major (weights W[N,K]); 1: B is [k][n] (ldb = k-row stride)
  int act;     // 0 none, 1 gelu(exact), 2 leaky 0.01
  float scale;
  int zdiv;    // z = zb*zdiv + zr
};

__device__ __forceinline__ float act_apply(float v, int act) {
  if (act == 1) return 0.5f * v * (1.0f + erff(v * 0.70710678118654752f));
  if (act == 2) return v > 0.0f ? v : 0.01f * v;
  return v;
}

__global__ __launch_bounds__(256) void k_gemm(GemmArgs g) {
  __shared__ __align__(16) _Float16 As[2][64 * 32];
  __shared__ __align__(16) _Float16 Bs[2][64 * 32];

  const int z  = blockIdx.z;
  const int zb = z / g.zdiv, zr = z % g.zdiv;
  const _Float16* A  = g.A + zb * g.zA0 + zr * g.zA1;
  const _Float16* Bp = g.B + zb * g.zB0 + zr * g.zB1;
  const long coff = zb * g.zC0 + zr * g.zC1;

  const int m0 = blockIdx.y * 64;
  const int n0 = blockIdx.x * 64;
  const int t    = threadIdx.x;
  const int lane = t & 31;
  const int w    = t >> 5;
  const int mw = (w & 3) * 16;    // 4 m-tiles of 16
  const int nw = (w >> 2) * 32;   // 2 n-slabs of 32

  float8 c0 = {}; float8 c1 = {};

  // tile-staging mapping: each thread moves one 16B chunk
  const int ldrow = t >> 2;        // 0..63
  const int ldk   = (t & 3) * 8;   // 0,8,16,24
  // bmode==1 mapping: contiguous-in-n load, scatter-transpose into LDS
  const int kr = t >> 3;           // 0..31
  const int nr = (t & 7) * 8;      // 0..56

  const _Float16* Asrc  = A  + (long)(m0 + ldrow) * g.lda + ldk;
  const _Float16* Bsrc0 = Bp + (long)(n0 + ldrow) * g.ldb + ldk;

  auto stage = [&](int ks, int buf) {
    const long k0 = (long)ks * 32;
    g2l_b128(Asrc + k0, &As[buf][ldrow * 32 + ldk]);
    if (g.bmode == 0) {
      g2l_b128(Bsrc0 + k0, &Bs[buf][ldrow * 32 + ldk]);
    } else {
      half8 v = *(const half8*)(Bp + (k0 + kr) * g.ldb + n0 + nr);
      #pragma unroll
      for (int i = 0; i < 8; ++i) Bs[buf][(nr + i) * 32 + kr] = v[i];
    }
  };

  const int nk = g.K / 32;
  stage(0, 0);
  for (int ks = 0; ks < nk; ++ks) {
    const int cur = ks & 1;
    if (ks + 1 < nk) stage(ks + 1, cur ^ 1);   // prefetch next tile (async)
    // wait for tile ks only; leave tile ks+1 in flight (async loads retire in order)
    if (ks + 1 < nk) { if (g.bmode == 0) wait_async<2>(); else wait_async<1>(); }
    else             wait_async<0>();
    __syncthreads();

    // --- fragments per ISA 16-bit layouts ---
    const _Float16* Ab = As[cur];
    const _Float16* Bb = Bs[cur];
    const int am = mw + (lane & 15);
    const int h8 = (lane >> 4) * 8;       // half-wave K phase
    half8 alo = *(const half8*)(Ab + am * 32 + h8);
    half8 ahi = *(const half8*)(Ab + am * 32 + 16 + h8);
    half16 a = __builtin_shufflevector(alo, ahi, 0,1,2,3,4,5,6,7,8,9,10,11,12,13,14,15);

    const int bn = nw + (lane & 15);
    half8 b0lo = *(const half8*)(Bb + bn * 32 + h8);
    half8 b0hi = *(const half8*)(Bb + bn * 32 + 16 + h8);
    half16 b0 = __builtin_shufflevector(b0lo, b0hi, 0,1,2,3,4,5,6,7,8,9,10,11,12,13,14,15);
    half8 b1lo = *(const half8*)(Bb + (bn + 16) * 32 + h8);
    half8 b1hi = *(const half8*)(Bb + (bn + 16) * 32 + 16 + h8);
    half16 b1 = __builtin_shufflevector(b1lo, b1hi, 0,1,2,3,4,5,6,7,8,9,10,11,12,13,14,15);

    c0 = __builtin_amdgcn_wmma_f32_16x16x32_f16(false, a, false, b0, (short)0, c0, false, false);
    c1 = __builtin_amdgcn_wmma_f32_16x16x32_f16(false, a, false, b1, (short)0, c1, false, false);

    __syncthreads();   // protect buffer being refilled next iteration
  }

  // --- epilogue: D layout VGPR r -> row = r + 8*(lane/16), col = lane%16 ---
  const int col0 = n0 + nw + (lane & 15);
  const int rowb = m0 + mw + (lane >> 4) * 8;
  #pragma unroll
  for (int r = 0; r < 8; ++r) {
    const int row = rowb + r;
    float v0 = c0[r] * g.scale;
    float v1 = c1[r] * g.scale;
    if (g.bias) { v0 += g.bias[col0]; v1 += g.bias[col0 + 16]; }
    v0 = act_apply(v0, g.act);
    v1 = act_apply(v1, g.act);
    const long o = coff + (long)row * g.ldc + col0;
    if (g.Cf) { g.Cf[o] = v0; g.Cf[o + 16] = v1; }
    else      { g.Ch[o] = (_Float16)v0; g.Ch[o + 16] = (_Float16)v1; }
  }
}

// --------------------------- small kernels ----------------------------------

__global__ __launch_bounds__(256) void k_f32h(const float* s, _Float16* d, long n) {
  long i = (long)blockIdx.x * 256 + threadIdx.x;
  if (i < n) d[i] = (_Float16)s[i];
}

__global__ __launch_bounds__(256) void k_padrows(const float* s, _Float16* d, int N, int K, int Kp) {
  long i = (long)blockIdx.x * 256 + threadIdx.x;
  if (i >= (long)N * Kp) return;
  int k = (int)(i % Kp); int n = (int)(i / Kp);
  d[i] = (k < K) ? (_Float16)s[(long)n * K + k] : (_Float16)0.0f;
}

__global__ __launch_bounds__(256) void k_add(float* x, const float* y, long n) {
  long i = (long)blockIdx.x * 256 + threadIdx.x;
  if (i < n) x[i] += y[i];
}

__global__ __launch_bounds__(256) void k_addpos(float* x, const float* p, long n, int mod) {
  long i = (long)blockIdx.x * 256 + threadIdx.x;
  if (i < n) x[i] += p[i % mod];
}

__global__ __launch_bounds__(256) void k_copy(const float* s, float* d, long n) {
  long i = (long)blockIdx.x * 256 + threadIdx.x;
  if (i < n) d[i] = s[i];
}

__global__ __launch_bounds__(256) void k_layernorm(const float* x, const float* g,
                                                   const float* b, _Float16* y, int rows) {
  int row  = blockIdx.x * 8 + (threadIdx.x >> 5);
  int lane = threadIdx.x & 31;
  if (row >= rows) return;
  const float* xr = x + (long)row * DIM;
  float v[12]; float s = 0.f, ss = 0.f;
  #pragma unroll
  for (int i = 0; i < 12; ++i) { v[i] = xr[lane + i * 32]; s += v[i]; ss += v[i] * v[i]; }
  #pragma unroll
  for (int o = 16; o > 0; o >>= 1) { s += __shfl_xor(s, o, 32); ss += __shfl_xor(ss, o, 32); }
  float mean = s * (1.0f / DIM);
  float var  = ss * (1.0f / DIM) - mean * mean;
  float inv  = rsqrtf(var + 1e-5f);
  #pragma unroll
  for (int i = 0; i < 12; ++i) {
    int c = lane + i * 32;
    y[(long)row * DIM + c] = (_Float16)((v[i] - mean) * inv * g[c] + b[c]);
  }
}

__global__ __launch_bounds__(256) void k_softmax(const float* s, _Float16* p, int rows, int n) {
  int row  = blockIdx.x * 8 + (threadIdx.x >> 5);
  int lane = threadIdx.x & 31;
  if (row >= rows) return;
  const float* sr = s + (long)row * n;
  int cnt = n >> 5;                 // 8 (256) or 14 (448)
  float v[14]; float mx = -3.0e38f;
  for (int i = 0; i < cnt; ++i) { v[i] = sr[lane + i * 32]; mx = fmaxf(mx, v[i]); }
  for (int o = 16; o > 0; o >>= 1) mx = fmaxf(mx, __shfl_xor(mx, o, 32));
  float sum = 0.f;
  for (int i = 0; i < cnt; ++i) { v[i] = expf(v[i] - mx); sum += v[i]; }
  for (int o = 16; o > 0; o >>= 1) sum += __shfl_xor(sum, o, 32);
  float inv = 1.0f / sum;
  for (int i = 0; i < cnt; ++i) p[(long)row * n + lane + i * 32] = (_Float16)(v[i] * inv);
}

__global__ __launch_bounds__(256) void k_im2col_chw(const float* src, _Float16* dst,
    int C, int H, int W, int kh, int stride, int pad, int OW, int OH, int Kp, int Keff) {
  long idx = (long)blockIdx.x * 256 + threadIdx.x;
  long total = (long)OH * OW * Kp;
  if (idx >= total) return;
  int k = (int)(idx % Kp); long p = idx / Kp;
  _Float16 v = (_Float16)0.0f;
  if (k < Keff) {
    int c = k / (kh * kh), r = k % (kh * kh), dy = r / kh, dx = r % kh;
    int oy = (int)(p / OW), ox = (int)(p % OW);
    int iy = oy * stride - pad + dy, ix = ox * stride - pad + dx;
    if (iy >= 0 && iy < H && ix >= 0 && ix < W)
      v = (_Float16)src[((long)c * H + iy) * W + ix];
  }
  dst[idx] = v;
}

__global__ __launch_bounds__(256) void k_im2col_hwc(const _Float16* src, _Float16* dst,
    int C, int H, int W, int stride, int pad, int OW, int OH) {
  int K = C * 9;
  long idx = (long)blockIdx.x * 256 + threadIdx.x;
  long total = (long)OH * OW * K;
  if (idx >= total) return;
  int k = (int)(idx % K); long p = idx / K;
  int c = k / 9, r = k % 9, dy = r / 3, dx = r % 3;
  int oy = (int)(p / OW), ox = (int)(p % OW);
  int iy = oy * stride - pad + dy, ix = ox * stride - pad + dx;
  _Float16 v = (_Float16)0.0f;
  if (iy >= 0 && iy < H && ix >= 0 && ix < W)
    v = src[((long)iy * W + ix) * C + c];
  dst[idx] = v;
}

__global__ __launch_bounds__(256) void k_pred(const _Float16* mln, const float* w2,
                                              const float* b2, float* pred, float* outpred) {
  int row  = blockIdx.x * 8 + (threadIdx.x >> 5);
  int lane = threadIdx.x & 31;
  float s = 0.f;
  #pragma unroll
  for (int i = 0; i < 12; ++i) {
    int c = lane + i * 32;
    s += (float)mln[(long)row * DIM + c] * w2[c];
  }
  #pragma unroll
  for (int o = 16; o > 0; o >>= 1) s += __shfl_xor(s, o, 32);
  if (lane == 0) { float v = s + b2[0]; pred[row] = v; outpred[row] = v; }
}

__global__ __launch_bounds__(256) void k_topk(const float* pred, int* tki, int* bki) {
  __shared__ float p[NT0];
  int b = blockIdx.x, i = threadIdx.x;
  p[i] = pred[b * NT0 + i];
  __syncthreads();
  float pi = p[i];
  int rd = 0, ra = 0;
  for (int j = 0; j < NT0; ++j) {
    float pj = p[j];
    bool tie = (pj == pi) && (j < i);
    if (pj > pi || tie) rd++;
    if (pj < pi || tie) ra++;
  }
  if (rd < KSPL) tki[b * KSPL + rd] = i;
  if (ra < KKEP) bki[b * KKEP + ra] = i;
}

__global__ __launch_bounds__(256) void k_gather_keep(const float* x0, const int* bki, float* x2) {
  long idx = (long)blockIdx.x * 256 + threadIdx.x;   // NB*KKEP*DIM
  int b = (int)(idx / (KKEP * DIM));
  int r = (int)(idx % (KKEP * DIM));
  int i = r / DIM, d = r % DIM;
  x2[(long)b * NT1 * DIM + (long)i * DIM + d] =
      x0[((long)b * NT0 + bki[b * KKEP + i]) * DIM + d];
}

__global__ __launch_bounds__(256) void k_gather_split(const float* x0, const int* tki, _Float16* sph) {
  long idx = (long)blockIdx.x * 256 + threadIdx.x;   // NB*KSPL*DIM
  int b = (int)(idx / (KSPL * DIM));
  int r = (int)(idx % (KSPL * DIM));
  int t = r / DIM, d = r % DIM;
  sph[idx] = (_Float16)x0[((long)b * NT0 + tki[b * KSPL + t]) * DIM + d];
}

__global__ __launch_bounds__(256) void k_feat_add(float* x2, const int* tki,
                                                  const float* rel, const float* se,
                                                  const float* hr) {
  long idx = (long)blockIdx.x * 256 + threadIdx.x;   // NB*256*DIM
  int b  = (int)(idx / (256 * DIM));
  int r  = (int)(idx % (256 * DIM));
  int tt = r / DIM, d = r % DIM;
  int ks = tt >> 2, q = tt & 3;
  int i  = tki[b * KSPL + ks];
  int y  = (i >> 4) * 2 + (q & 1);
  int x  = (i & 15) * 2 + (q >> 1);
  x2[(long)b * NT1 * DIM + (long)(KKEP + tt) * DIM + d] +=
      rel[q * DIM + d] + se[d] + hr[((long)b * 1024 + y * 32 + x) * DIM + d];
}

__global__ __launch_bounds__(256) void k_coords(const int* tki, const int* bki, float* out) {
  int idx = blockIdx.x * 256 + threadIdx.x;
  if (idx >= NB * NT1) return;
  int b = idx / NT1, i = idx % NT1;
  float c0, c1, c2;
  if (i < KKEP) {
    int s = bki[b * KKEP + i];
    c0 = 0.f; c1 = (float)((s >> 4) * 2); c2 = (float)((s & 15) * 2);
  } else {
    int t = (i - KKEP) >> 2, q = (i - KKEP) & 3;
    int s = tki[b * KSPL + t];
    c0 = 1.f;
    c1 = (float)((s >> 4) * 2 + (q & 1));
    c2 = (float)((s & 15) * 2 + (q >> 1));
  }
  out[(long)idx * 3 + 0] = c0;
  out[(long)idx * 3 + 1] = c1;
  out[(long)idx * 3 + 2] = c2;
}

__global__ __launch_bounds__(256) void k_meta(float* out) {
  int b = blockIdx.x, i = threadIdx.x;
  out[((long)b * NT0 + i) * 2 + 0] = (float)((i >> 4) * 2);
  out[((long)b * NT0 + i) * 2 + 1] = (float)((i & 15) * 2);
}

// ------------------------------ host ----------------------------------------

extern "C" void kernel_launch(void* const* d_in, const int* in_sizes, int n_in,
                              void* d_out, int out_size, void* d_ws, size_t ws_size,
                              hipStream_t stream) {
  (void)in_sizes; (void)n_in; (void)out_size; (void)ws_size;
  auto IN = [&](int i) { return (const float*)d_in[i]; };

  size_t off = 0;
  auto A16 = [&](size_t e) { _Float16* p = (_Float16*)((char*)d_ws + off);
                             off = (off + e * 2 + 255) & ~(size_t)255; return p; };
  auto A32 = [&](size_t e) { float* p = (float*)((char*)d_ws + off);
                             off = (off + e * 4 + 255) & ~(size_t)255; return p; };
  auto AI  = [&](size_t e) { int* p = (int*)((char*)d_ws + off);
                             off = (off + e * 4 + 255) & ~(size_t)255; return p; };

  auto cvt = [&](const float* s, _Float16* d, long n) {
    k_f32h<<<(unsigned)((n + 255) / 256), 256, 0, stream>>>(s, d, n);
  };

  auto gemm = [&](const _Float16* A, long lda, long zA0, long zA1,
                  const _Float16* Bm, long ldb, int bmode, long zB0, long zB1,
                  const float* bias, float* Cf, _Float16* Ch, long ldc, long zC0, long zC1,
                  int M, int N, int K, int act, float scale, int Z, int zdiv) {
    GemmArgs g{A, lda, zA0, zA1, Bm, ldb, zB0, zB1, bias,
               Cf, Ch, ldc, zC0, zC1, M, N, K, bmode, act, scale, zdiv};
    dim3 gr((unsigned)(N / 64), (unsigned)(M / 64), (unsigned)Z);
    k_gemm<<<gr, 256, 0, stream>>>(g);
  };

  // ---- input index map (insertion-order flattening of setup_inputs) ----
  // 0: im; 1..8: pe_convs[i].{w,b}; 9: pos_embed; 10: rel_pos; 11: scale_emb;
  // 12..107: layers (8 blocks x [ln1.g,ln1.b,qkv.w,qkv.b,proj.w,proj.b,
  //                              ln2.g,ln2.b,fc1.w,fc1.b,fc2.w,fc2.b]);
  // 108,109: down; 110,111: split; 112,113: ml_lin1; 114,115: ml_ln;
  // 116,117: ml_lin2; 118,119: hrp.
  const float* im = IN(0);

  // ---- convert weights to f16 ----
  _Float16* w1h = A16((size_t)DIM * 32);
  k_padrows<<<(DIM * 32 + 255) / 256, 256, 0, stream>>>(IN(1), w1h, DIM, 27, 32);
  _Float16* wc2 = A16((size_t)DIM * 3456); cvt(IN(3), wc2, (long)DIM * 3456);
  _Float16* wc3 = A16((size_t)DIM * 3456); cvt(IN(5), wc3, (long)DIM * 3456);
  _Float16* wc4 = A16((size_t)DIM * 3456); cvt(IN(7), wc4, (long)DIM * 3456);
  _Float16* hrph = A16((size_t)DIM * 192); cvt(IN(118), hrph, (long)DIM * 192);

  _Float16 *wq[8], *wp[8], *wf1[8], *wf2[8];
  for (int i = 0; i < 8; ++i) {
    int pb = 12 + i * 12;
    wq[i]  = A16((size_t)1152 * DIM); cvt(IN(pb + 2),  wq[i],  (long)1152 * DIM);
    wp[i]  = A16((size_t)DIM * DIM);  cvt(IN(pb + 4),  wp[i],  (long)DIM * DIM);
    wf1[i] = A16((size_t)1536 * DIM); cvt(IN(pb + 8),  wf1[i], (long)1536 * DIM);
    wf2[i] = A16((size_t)DIM * 1536); cvt(IN(pb + 10), wf2[i], (long)DIM * 1536);
  }
  _Float16* downw = A16((size_t)DIM * DIM);  cvt(IN(108), downw, (long)DIM * DIM);
  _Float16* splitw = A16((size_t)1536 * DIM); cvt(IN(110), splitw, (long)1536 * DIM);
  _Float16* ml1w = A16((size_t)DIM * DIM);   cvt(IN(112), ml1w, (long)DIM * DIM);

  // ---- activation buffers ----
  _Float16* Abuf = A16((size_t)4096 * 3456);
  _Float16* actb = A16((size_t)16384 * DIM);
  float* x0   = A32((size_t)NB * NT0 * DIM);
  _Float16* lnh  = A16((size_t)3584 * DIM);
  _Float16* xh   = A16((size_t)3584 * DIM);
  float* qkvf = A32((size_t)3584 * 1152);
  _Float16* qkvh = A16((size_t)3584 * 1152);
  float* scores = A32((size_t)NB * NHD * NT1 * NT1);
  _Float16* ph = A16((size_t)NB * NHD * NT1 * NT1);
  float* attnf = A32((size_t)3584 * DIM);
  _Float16* attnh = A16((size_t)3584 * DIM);
  _Float16* h1h = A16((size_t)3584 * 1536);
  float* tmpf = A32((size_t)3584 * DIM);
  float* mbuf = A32((size_t)NB * NT0 * DIM);
  _Float16* mlnh = A16((size_t)NB * NT0 * DIM);
  float* predb = A32(NB * NT0);
  int* tki = AI(NB * KSPL);
  int* bki = AI(NB * KKEP);
  _Float16* sph = A16((size_t)NB * KSPL * DIM);
  float* hrbuf = A32((size_t)NB * 1024 * DIM);
  float* x2 = A32((size_t)NB * NT1 * DIM);
  float* x3 = A32((size_t)NB * NT1 * DIM);

  // ---- patch-embed conv chain (per batch, im2col + WMMA GEMM, GELU fused) ----
  for (int b = 0; b < NB; ++b) {
    const float* imb = im + (long)b * 3 * 256 * 256;
    // conv1: 3->384, 256->128
    k_im2col_chw<<<(16384 * 32) / 256, 256, 0, stream>>>(imb, Abuf, 3, 256, 256, 3, 2, 1, 128, 128, 32, 27);
    gemm(Abuf, 32, 0, 0, w1h, 32, 0, 0, 0, IN(2), nullptr, actb, DIM, 0, 0,
         16384, DIM, 32, 1, 1.f, 1, 1);
    // conv2: 384->384, 128->64
    k_im2col_hwc<<<(unsigned)(((long)4096 * 3456) / 256), 256, 0, stream>>>(actb, Abuf, DIM, 128, 128, 2, 1, 64, 64);
    gemm(Abuf, 3456, 0, 0, wc2, 3456, 0, 0, 0, IN(4), nullptr, actb, DIM, 0, 0,
         4096, DIM, 3456, 1, 1.f, 1, 1);
    // conv3: 64->32
    k_im2col_hwc<<<(unsigned)(((long)1024 * 3456) / 256), 256, 0, stream>>>(actb, Abuf, DIM, 64, 64, 2, 1, 32, 32);
    gemm(Abuf, 3456, 0, 0, wc3, 3456, 0, 0, 0, IN(6), nullptr, actb, DIM, 0, 0,
         1024, DIM, 3456, 1, 1.f, 1, 1);
    // conv4: 32->16 -> tokens [256, 384] f32 directly
    k_im2col_hwc<<<(unsigned)(((long)256 * 3456) / 256), 256, 0, stream>>>(actb, Abuf, DIM, 32, 32, 2, 1, 16, 16);
    gemm(Abuf, 3456, 0, 0, wc4, 3456, 0, 0, 0, IN(8),
         x0 + (long)b * NT0 * DIM, nullptr, DIM, 0, 0, 256, DIM, 3456, 1, 1.f, 1, 1);
    // hr patches: 8x8 stride 8 -> [1024, 384]
    k_im2col_chw<<<(1024 * 192) / 256, 256, 0, stream>>>(imb, Abuf, 3, 256, 256, 8, 8, 0, 32, 32, 192, 192);
    gemm(Abuf, 192, 0, 0, hrph, 192, 0, 0, 0, IN(119),
         hrbuf + (long)b * 1024 * DIM, nullptr, DIM, 0, 0, 1024, DIM, 192, 0, 1.f, 1, 1);
  }
  // x += pos_embed
  k_addpos<<<(NB * NT0 * DIM) / 256, 256, 0, stream>>>(x0, IN(9), (long)NB * NT0 * DIM, NT0 * DIM);

  // ---- transformer block (WMMA GEMMs + batched attention) ----
  auto run_block = [&](float* x, int M, int Nt, int bi) {
    int pb = 12 + bi * 12;
    long n384 = (long)M * DIM;
    // ln1 -> f16
    k_layernorm<<<M / 8, 256, 0, stream>>>(x, IN(pb + 0), IN(pb + 1), lnh, M);
    // qkv
    gemm(lnh, DIM, 0, 0, wq[bi], DIM, 0, 0, 0, IN(pb + 3), qkvf, nullptr, 1152, 0, 0,
         M, 1152, DIM, 0, 1.f, 1, 1);
    cvt(qkvf, qkvh, (long)M * 1152);
    // scores = Q K^T / 8   (z = b*NH + h)
    gemm(qkvh, 1152, (long)Nt * 1152, 64,
         qkvh + DIM, 1152, 0, (long)Nt * 1152, 64, nullptr,
         scores, nullptr, Nt, (long)NHD * Nt * Nt, (long)Nt * Nt,
         Nt, Nt, 64, 0, 0.125f, NB * NHD, NHD);
    k_softmax<<<(NB * NHD * Nt) / 8, 256, 0, stream>>>(scores, ph, NB * NHD * Nt, Nt);
    // out = P V
    gemm(ph, Nt, (long)NHD * Nt * Nt, (long)Nt * Nt,
         qkvh + 2 * DIM, 1152, 1, (long)Nt * 1152, 64, nullptr,
         attnf, nullptr, DIM, (long)Nt * DIM, 64,
         Nt, 64, Nt, 0, 1.f, NB * NHD, NHD);
    cvt(attnf, attnh, n384);
    gemm(attnh, DIM, 0, 0, wp[bi], DIM, 0, 0, 0, IN(pb + 5), tmpf, nullptr, DIM, 0, 0,
         M, DIM, DIM, 0, 1.f, 1, 1);
    k_add<<<(unsigned)(n384 / 256), 256, 0, stream>>>(x, tmpf, n384);
    // mlp
    k_layernorm<<<M / 8, 256, 0, stream>>>(x, IN(pb + 6), IN(pb + 7), lnh, M);
    gemm(lnh, DIM, 0, 0, wf1[bi], DIM, 0, 0, 0, IN(pb + 9), nullptr, h1h, 1536, 0, 0,
         M, 1536, DIM, 1, 1.f, 1, 1);
    gemm(h1h, 1536, 0, 0, wf2[bi], 1536, 0, 0, 0, IN(pb + 11), tmpf, nullptr, DIM, 0, 0,
         M, DIM, 1536, 0, 1.f, 1, 1);
    k_add<<<(unsigned)(n384 / 256), 256, 0, stream>>>(x, tmpf, n384);
  };

  for (int bi = 0; bi < 4; ++bi) run_block(x0, NB * NT0, NT0, bi);

  // ---- meta head + routing ----
  cvt(x0, xh, (long)NB * NT0 * DIM);
  gemm(xh, DIM, 0, 0, ml1w, DIM, 0, 0, 0, IN(113), mbuf, nullptr, DIM, 0, 0,
       NB * NT0, DIM, DIM, 2, 1.f, 1, 1);
  k_layernorm<<<(NB * NT0) / 8, 256, 0, stream>>>(mbuf, IN(114), IN(115), mlnh, NB * NT0);
  k_pred<<<(NB * NT0) / 8, 256, 0, stream>>>(mlnh, IN(116), IN(117), predb,
                                             (float*)d_out + OFF_PRED);
  k_topk<<<NB, 256, 0, stream>>>(predb, tki, bki);

  k_gather_keep<<<(NB * KKEP * DIM) / 256, 256, 0, stream>>>(x0, bki, x2);
  k_gather_split<<<(NB * KSPL * DIM) / 256, 256, 0, stream>>>(x0, tki, sph);
  // split linear: [64,384] x [1536,384]^T per batch, written into x2 rows 192..447
  gemm(sph, DIM, (long)KSPL * DIM, 0, splitw, DIM, 0, 0, 0, IN(111),
       x2 + (long)KKEP * DIM, nullptr, 1536, (long)NT1 * DIM, 0,
       KSPL, 1536, DIM, 0, 1.f, NB, 1);
  k_feat_add<<<(NB * 256 * DIM) / 256, 256, 0, stream>>>(x2, tki, IN(10), IN(11), hrbuf);
  k_coords<<<(NB * NT1 + 255) / 256, 256, 0, stream>>>(tki, bki, (float*)d_out + OFF_COORDS);
  k_meta<<<NB, 256, 0, stream>>>((float*)d_out + OFF_META);

  // ---- down + stage-1 blocks ----
  cvt(x2, xh, (long)NB * NT1 * DIM);
  gemm(xh, DIM, 0, 0, downw, DIM, 0, 0, 0, IN(109), x3, nullptr, DIM, 0, 0,
       NB * NT1, DIM, DIM, 0, 1.f, 1, 1);
  for (int bi = 4; bi < 8; ++bi) run_block(x3, NB * NT1, NT1, bi);

  k_copy<<<(CNT_X) / 256, 256, 0, stream>>>(x3, (float*)d_out + OFF_X, (long)CNT_X);
}
